// SOCA_44865228374074
// MI455X (gfx1250) — matrix operand order repaired and checked by reading, not compiled, not run on hardware
//
#include <hip/hip_runtime.h>
#include <hip/hip_bf16.h>
#include <math.h>

// ---------------------------------------------------------------------------
// SOCA forward for MI455X (gfx1250, wave32, WMMA).
//   B=16, C=128, H=W=192, M=H*W=36864, RED=8 (hidden 16), ITERN=5
//   - Gram/covariance: v_wmma_f32_16x16x32_bf16, bf16 staged ONCE in LDS,
//     fragments read as ds_load_b128, globals software-pipelined.
//   - Newton-Schulz 128x128 matmuls: v_wmma_f32_16x16x4_f32, LDS-resident.
// ---------------------------------------------------------------------------

typedef __attribute__((ext_vector_type(16))) __bf16 v16bf;
typedef __attribute__((ext_vector_type(8)))  __bf16 v8bf;
typedef __attribute__((ext_vector_type(4)))  __bf16 v4bf;
typedef __attribute__((ext_vector_type(8)))  float  v8f;
typedef __attribute__((ext_vector_type(2)))  float  v2f;

union bf16x16 { v16bf v; v8bf h[2]; };

constexpr int B_  = 16;
constexpr int C_  = 128;
constexpr int M_  = 192 * 192;        // 36864
constexpr int NCHUNK = 16;            // m-chunks per batch
constexpr int CHUNK  = M_ / NCHUNK;   // 2304 (divisible by 32)
constexpr int KSTEP  = 32;            // bf16 WMMA K
constexpr int LDSROW = 40;            // bf16 row stride (80B: 16B-aligned, bank-spread)

// ---------------------------------------------------------------------------
// Kernel 1: partial Gram G = X @ X^T over an m-chunk, plus per-channel sums.
// grid = (NCHUNK, B), block = 256 (8 waves). Wave w owns output tile-row I=w.
// ---------------------------------------------------------------------------
__global__ __launch_bounds__(256) void k_gram(const float* __restrict__ x,
                                              float* __restrict__ pg,
                                              float* __restrict__ csp) {
  const int chunk = blockIdx.x;
  const int b     = blockIdx.y;
  const int t     = threadIdx.x;
  const int wave  = t >> 5;
  const int lane  = t & 31;
  const int half  = lane >> 4;
  const int l16   = lane & 15;

  __shared__ __bf16 sxb[C_ * LDSROW];              // 128 ch x 32 m (pad to 40) = 10KB
  __shared__ float  csred[4 * 256];                // channel-sum reduction

  const float* xb = x + (size_t)b * C_ * M_;
  const int m0 = chunk * CHUNK;

  // Thread t owns 4 fixed (channel, m-offset) slots per k-step:
  //   it in 0..3 : ch = (t>>3) + it*32 , mo = (t&7)*4
  const int chBase = t >> 3;
  const int mo     = (t & 7) * 4;

  v8f acc[8];
  for (int j = 0; j < 8; ++j)
    for (int e = 0; e < 8; ++e) acc[j][e] = 0.f;
  float csum[4] = {0.f, 0.f, 0.f, 0.f};

  // Prologue: load first k-step tile into registers.
  float4 r[4];
  #pragma unroll
  for (int it = 0; it < 4; ++it)
    r[it] = *(const float4*)(xb + (size_t)(chBase + it * 32) * M_ + m0 + mo);

  for (int k0 = 0; k0 < CHUNK; k0 += KSTEP) {
    // Store current tile to LDS as bf16; accumulate f32 channel sums.
    #pragma unroll
    for (int it = 0; it < 4; ++it) {
      float4 v = r[it];
      csum[it] += v.x + v.y + v.z + v.w;
      v4bf p;
      p[0] = (__bf16)v.x; p[1] = (__bf16)v.y;
      p[2] = (__bf16)v.z; p[3] = (__bf16)v.w;
      *(v4bf*)(sxb + (chBase + it * 32) * LDSROW + mo) = p;
    }
    __syncthreads();

    // Software pipeline: issue next k-step's global loads now.
    if (k0 + KSTEP < CHUNK) {
      #pragma unroll
      for (int it = 0; it < 4; ++it)
        r[it] = *(const float4*)(xb + (size_t)(chBase + it * 32) * M_ +
                                 m0 + k0 + KSTEP + mo);
    }

    // A fragment (16x32 bf16, ISA 7.12.2):
    //   lane<16 : row=l16, elems 0..7 -> K0..7,  8..15 -> K16..23
    //   lane>=16: row=l16, elems 0..7 -> K8..15, 8..15 -> K24..31
    bf16x16 a;
    {
      const __bf16* rowp = sxb + (wave * 16 + l16) * LDSROW;
      a.h[0] = *(const v8bf*)(rowp + half * 8);
      a.h[1] = *(const v8bf*)(rowp + 16 + half * 8);
    }
    // B fragment (32x16): col=l16, same K order as A (elems 0..7 -> K8h..8h+7, 8..15 -> K16+8h..).
    #pragma unroll
    for (int J = 0; J < 8; ++J) {
      bf16x16 bb;
      const __bf16* colp = sxb + (J * 16 + l16) * LDSROW + half * 8;
      bb.h[0] = *(const v8bf*)(colp);
      bb.h[1] = *(const v8bf*)(colp + 16);
      acc[J] = __builtin_amdgcn_wmma_f32_16x16x32_bf16(
          false, a.v, false, bb.v, (short)0, acc[J], false, false);
    }
    __syncthreads();
  }

  // Write partial Gram. C/D layout: vgpr v, lane<16 -> row v, lane>=16 -> row 8+v.
  float* pgb = pg + (size_t)(b * NCHUNK + chunk) * (C_ * C_);
  #pragma unroll
  for (int J = 0; J < 8; ++J)
    #pragma unroll
    for (int v = 0; v < 8; ++v)
      pgb[(wave * 16 + half * 8 + v) * C_ + J * 16 + l16] = acc[J][v];

  // Reduce per-channel sums: channel c <- threads (c&31)*8..+7 of group c>>5.
  #pragma unroll
  for (int it = 0; it < 4; ++it) csred[it * 256 + t] = csum[it];
  __syncthreads();
  if (t < C_) {
    const float* src = csred + (t >> 5) * 256 + (t & 31) * 8;
    float s = 0.f;
    #pragma unroll
    for (int u = 0; u < 8; ++u) s += src[u];
    csp[(b * NCHUNK + chunk) * C_ + t] = s;
  }
}

// ---------------------------------------------------------------------------
// Kernel 2: reduce partials -> cov = G/M - mu mu^T ; normA = sum(cov);
// A = cov / normA. grid = B, block = 256.
// ---------------------------------------------------------------------------
__global__ __launch_bounds__(256) void k_cov(const float* __restrict__ pg,
                                             const float* __restrict__ csp,
                                             float* __restrict__ covA,
                                             float* __restrict__ norms) {
  const int b = blockIdx.x;
  const int t = threadIdx.x;
  __shared__ float mu[C_];
  __shared__ float red[256];

  if (t < C_) {
    float s = 0.f;
    for (int ck = 0; ck < NCHUNK; ++ck) s += csp[(b * NCHUNK + ck) * C_ + t];
    mu[t] = s / (float)M_;
  }
  __syncthreads();

  float* cb = covA + (size_t)b * C_ * C_;
  float lsum = 0.f;
  for (int i = 0; i < 64; ++i) {
    int e = t + i * 256;
    float g = 0.f;
    for (int ck = 0; ck < NCHUNK; ++ck)
      g += pg[(size_t)(b * NCHUNK + ck) * (C_ * C_) + e];
    int row = e >> 7, col = e & 127;
    float cov = g / (float)M_ - mu[row] * mu[col];
    cb[e] = cov;
    lsum += cov;
  }
  red[t] = lsum;
  __syncthreads();
  for (int s = 128; s > 0; s >>= 1) {
    if (t < s) red[t] += red[t + s];
    __syncthreads();
  }
  float normA = red[0];
  if (t == 0) { norms[b * 2] = normA; norms[b * 2 + 1] = sqrtf(normA); }
  float inv = 1.f / normA;
  for (int i = 0; i < 64; ++i) {
    int e = t + i * 256;
    cb[e] *= inv;                                   // A = cov / normA
  }
}

// ---------------------------------------------------------------------------
// 128x128x128 f32 matmul from LDS via v_wmma_f32_16x16x4_f32.
// Wave `wave` computes tile-row I=wave, all J. Caller handles barriers.
// ---------------------------------------------------------------------------
__device__ __forceinline__ void mm128(float* __restrict__ D,
                                      const float* __restrict__ S1,
                                      const float* __restrict__ S2,
                                      int wave, int lane) {
  const int half = lane >> 4;
  const int l16  = lane & 15;
  v8f acc[8];
  for (int j = 0; j < 8; ++j)
    for (int e = 0; e < 8; ++e) acc[j][e] = 0.f;

  for (int k0 = 0; k0 < C_; k0 += 4) {
    const int kb = k0 + half * 2;                   // lane<16 -> K0,1 ; else K2,3
    v2f a;
    a.x = S1[(wave * 16 + l16) * C_ + kb];
    a.y = S1[(wave * 16 + l16) * C_ + kb + 1];
    #pragma unroll
    for (int J = 0; J < 8; ++J) {
      v2f bb;
      bb.x = S2[kb * C_ + J * 16 + l16];
      bb.y = S2[(kb + 1) * C_ + J * 16 + l16];
      acc[J] = __builtin_amdgcn_wmma_f32_16x16x4_f32(
          false, a, false, bb, (short)0, acc[J], false, false);
    }
  }
  #pragma unroll
  for (int J = 0; J < 8; ++J)
    #pragma unroll
    for (int v = 0; v < 8; ++v)
      D[(wave * 16 + half * 8 + v) * C_ + J * 16 + l16] = acc[J][v];
}

// ---------------------------------------------------------------------------
// Kernel 3: Newton-Schulz (ITERN=5) fully in LDS (4 x 64KB buffers = 256KB of
// CDNA5's 320KB/WGP), then column-mean -> SE MLP -> per-(b,c) gate.
// grid = B, block = 256, dynamic LDS = 256KB.
// ---------------------------------------------------------------------------
__global__ __launch_bounds__(256) void k_ns(const float* __restrict__ covA,
                                            const float* __restrict__ norms,
                                            const float* __restrict__ w1,
                                            const float* __restrict__ b1,
                                            const float* __restrict__ w2,
                                            const float* __restrict__ b2,
                                            float* __restrict__ gate) {
  extern __shared__ float lds[];
  float* buf0 = lds;
  float* buf1 = lds + 16384;
  float* buf2 = lds + 2 * 16384;
  float* buf3 = lds + 3 * 16384;
  __shared__ float svec[C_];
  __shared__ float hvec[16];

  const int b = blockIdx.x;
  const int t = threadIdx.x;
  const int wave = t >> 5, lane = t & 31;
  const float* Ab = covA + (size_t)b * C_ * C_;

  // U(buf3) = A ; Z(buf1) = ZY0 = 0.5*(3I - A)
  for (int i = 0; i < 64; ++i) {
    int e = t + i * 256;
    float a = Ab[e];
    buf3[e] = a;
    int row = e >> 7, col = e & 127;
    buf1[e] = 0.5f * ((row == col ? 3.f : 0.f) - a);
  }
  __syncthreads();
  mm128(buf0, buf3, buf1, wave, lane);              // Y = A @ ZY0
  __syncthreads();

  float *pY = buf0, *pZ = buf1, *pT = buf2, *pU = buf3;
  for (int it = 1; it < 4; ++it) {                  // range(1, ITERN-1)
    mm128(pU, pZ, pY, wave, lane);                  // U = Z@Y
    __syncthreads();
    for (int i = 0; i < 64; ++i) {
      int e = t + i * 256;
      int row = e >> 7, col = e & 127;
      pU[e] = 0.5f * ((row == col ? 3.f : 0.f) - pU[e]);   // ZY
    }
    __syncthreads();
    mm128(pT, pY, pU, wave, lane);                  // newY = Y@ZY
    __syncthreads();
    mm128(pY, pU, pZ, wave, lane);                  // newZ = ZY@Z (into old Y buf)
    __syncthreads();
    float* freed = pZ;
    pZ = pY; pY = pT; pT = freed;                   // pU stays scratch
  }

  // Final: R = (Y @ (3I - Z)) @ Y ; sqrt_cov = 0.5*sqrt(normA)*R
  for (int i = 0; i < 64; ++i) {
    int e = t + i * 256;
    int row = e >> 7, col = e & 127;
    pU[e] = (row == col ? 3.f : 0.f) - pZ[e];
  }
  __syncthreads();
  mm128(pT, pY, pU, wave, lane);
  __syncthreads();
  mm128(pU, pT, pY, wave, lane);                    // R in pU
  __syncthreads();

  const float sn = norms[b * 2 + 1];                // sqrt(normA)
  if (t < C_) {
    float s = 0.f;
    for (int r = 0; r < C_; ++r) s += pU[r * C_ + t];
    svec[t] = s * (0.5f * sn / (float)C_);          // mean over rows (axis=1)
  }
  __syncthreads();
  if (t < 16) {
    float h = b1[t];
    for (int c = 0; c < C_; ++c) h += svec[c] * w1[t * C_ + c];
    hvec[t] = h > 0.f ? h : 0.f;
  }
  __syncthreads();
  if (t < C_) {
    float g = b2[t];
    for (int j = 0; j < 16; ++j) g += hvec[j] * w2[t * 16 + j];
    gate[b * C_ + t] = 1.f / (1.f + expf(-g));
  }
}

// ---------------------------------------------------------------------------
// Kernel 4: out = gate[b,c] * x   (float4, grid-stride)
// ---------------------------------------------------------------------------
__global__ __launch_bounds__(256) void k_scale(const float* __restrict__ x,
                                               const float* __restrict__ gate,
                                               float* __restrict__ out) {
  const long long n4 = (long long)B_ * C_ * M_ / 4;
  const int M4 = M_ / 4;                            // 9216
  for (long long i = (long long)blockIdx.x * blockDim.x + threadIdx.x; i < n4;
       i += (long long)gridDim.x * blockDim.x) {
    long long cm = i / M4;                          // == b*C + c
    float g = gate[cm];
    float4 v = ((const float4*)x)[i];
    v.x *= g; v.y *= g; v.z *= g; v.w *= g;
    ((float4*)out)[i] = v;
  }
}

// ---------------------------------------------------------------------------
extern "C" void kernel_launch(void* const* d_in, const int* in_sizes, int n_in,
                              void* d_out, int out_size, void* d_ws, size_t ws_size,
                              hipStream_t stream) {
  const float* x  = (const float*)d_in[0];
  const float* w1 = (const float*)d_in[1];
  const float* b1 = (const float*)d_in[2];
  const float* w2 = (const float*)d_in[3];
  const float* b2 = (const float*)d_in[4];
  float* out = (float*)d_out;

  // Workspace layout (floats): ~18 MB total, fully rewritten every call.
  float* ws    = (float*)d_ws;
  float* pg    = ws;                                          // B*NCHUNK*128*128
  float* csp   = pg  + (size_t)B_ * NCHUNK * C_ * C_;         // B*NCHUNK*128
  float* covA  = csp + (size_t)B_ * NCHUNK * C_;              // B*128*128
  float* norms = covA + (size_t)B_ * C_ * C_;                 // 2*B
  float* gate  = norms + 2 * B_;                              // B*128

  dim3 g1(NCHUNK, B_);
  k_gram<<<g1, 256, 0, stream>>>(x, pg, csp);
  k_cov<<<B_, 256, 0, stream>>>(pg, csp, covA, norms);
  k_ns<<<B_, 256, 4 * C_ * C_ * sizeof(float), stream>>>(covA, norms, w1, b1,
                                                         w2, b2, gate);
  k_scale<<<4096, 256, 0, stream>>>(x, gate, out);
}
